// Attention_73770358276185
// MI455X (gfx1250) — matrix operand order repaired and checked by reading, not compiled, not run on hardware
//
#include <hip/hip_runtime.h>

// ---------------------------------------------------------------------------
// MI455X (gfx1250) fused conv-bn-gelu + flash attention
//   f,g,h projections -> bf16 workspace
//   S = g^T f (K=64), online softmax (fp32), O = P h^T (K=N), out = gamma*O^T + q
//   Matrix math: v_wmma_f32_16x16x32_bf16 (wave32, fp32 accumulate)
//   + global_prefetch_b8 hints for next-tile staging (L2 -> WGP latency hiding)
// ---------------------------------------------------------------------------

typedef __attribute__((ext_vector_type(16))) __bf16 bf16x16;
typedef __attribute__((ext_vector_type(8)))  __bf16 bf16x8;
typedef __attribute__((ext_vector_type(8)))  float  f32x8;

#define N_SEQ 4096
#define DIM1  256
#define DIM2  128
#define LAYER 64
#define BATCH 8

__device__ __forceinline__ float gelu_exact(float x) {
    return 0.5f * x * (1.0f + erff(x * 0.70710678118654752f));
}

// ---------------- Kernel 1: 1x1 conv + folded BN + exact GELU -> bf16 -------
// grid: ((Cout/4) * (N/256), B), block 256. Each thread computes 4 output rows
// (stride Cout/4) for one n, reusing each x read 4x; x reads are coalesced.
__global__ __launch_bounds__(256) void convbngelu_kernel(
    const float* __restrict__ x, const float* __restrict__ W,
    const float* __restrict__ bconv, const float* __restrict__ bn_s,
    const float* __restrict__ bn_b, const float* __restrict__ bn_m,
    const float* __restrict__ bn_v, __bf16* __restrict__ out,
    int Cin, int Cout)
{
    const int b  = blockIdx.y;
    const int og = blockIdx.x >> 4;                       // 16 n-blocks per og
    const int n  = ((blockIdx.x & 15) << 8) + threadIdx.x;
    const int ostep = Cout >> 2;

    const float* xb = x + ((size_t)b * Cin) * N_SEQ + n;
    const float* w0 = W + (size_t)(og            ) * Cin;
    const float* w1 = W + (size_t)(og +     ostep) * Cin;
    const float* w2 = W + (size_t)(og + 2 * ostep) * Cin;
    const float* w3 = W + (size_t)(og + 3 * ostep) * Cin;

    float a0 = 0.f, a1 = 0.f, a2 = 0.f, a3 = 0.f;
    for (int c = 0; c < Cin; ++c) {
        float xv = xb[(size_t)c * N_SEQ];
        a0 += w0[c] * xv; a1 += w1[c] * xv; a2 += w2[c] * xv; a3 += w3[c] * xv;
    }
    float accs[4] = {a0, a1, a2, a3};
#pragma unroll
    for (int j = 0; j < 4; ++j) {
        int   o   = og + j * ostep;
        float inv = bn_s[o] * rsqrtf(bn_v[o] + 1e-5f);
        float y   = (accs[j] + bconv[o] - bn_m[o]) * inv + bn_b[o];
        out[((size_t)b * Cout + o) * N_SEQ + n] = (__bf16)gelu_exact(y);
    }
}

// ---------------- WMMA helpers ---------------------------------------------
__device__ __forceinline__ f32x8 wmma_bf16(bf16x16 a, bf16x16 b, f32x8 c) {
    return __builtin_amdgcn_wmma_f32_16x16x32_bf16(
        /*neg_a=*/false, a, /*neg_b=*/false, b,
        /*c_mod=*/(short)0, c, /*reuse_a=*/false, /*reuse_b=*/false);
}

// A fragment (16x32 bf16). Lane L holds M=L&15,
// element i maps to K = (L>=16 ? 16 : 0) + i (matches B fragments).
// With row-major storage this is one contiguous 32B chunk.
__device__ __forceinline__ bf16x16 ld_afrag(const __bf16* row, int kb, int hi8) {
    return *(const bf16x16*)(row + kb + 2 * hi8);
}

// reductions within each 16-lane half (rows of one C-layout half)
__device__ __forceinline__ float redmax16(float v) {
#pragma unroll
    for (int m = 1; m < 16; m <<= 1) v = fmaxf(v, __shfl_xor(v, m, 32));
    return v;
}
__device__ __forceinline__ float redsum16(float v) {
#pragma unroll
    for (int m = 1; m < 16; m <<= 1) v += __shfl_xor(v, m, 32);
    return v;
}

// ---------------- Kernel 2: flash attention --------------------------------
// grid: (N/128, B), block 256 (8 waves). Wave w owns rows [w*16, w*16+16).
__global__ __launch_bounds__(256, 1) void flash_attn_kernel(
    const __bf16* __restrict__ fmat,   // [B,64,N]  keys   (S columns)
    const __bf16* __restrict__ gmat,   // [B,64,N]  queries(S rows)
    const __bf16* __restrict__ hmat,   // [B,256,N] values
    const float*  __restrict__ qres,   // residual [B,256,N]
    const float*  __restrict__ gamma,
    float* __restrict__ outp)          // [B,256,N]
{
    __shared__ __align__(32) __bf16 sQ [128 * 64];   // [i][l]  16 KB
    __shared__ __align__(32) __bf16 sKV[128 * 64];   // K:[j][l] / V:[c][j] 16 KB
    __shared__ __align__(32) __bf16 sP [128 * 128];  // [i][j]  32 KB

    const int tid  = threadIdx.x;
    const int lane = tid & 31;
    const int w    = tid >> 5;
    const int l4   = lane & 15;
    const int hi16 = lane & 16;      // 0 / 16 (B-frag, C-col half)
    const int hi8  = hi16 >> 1;      // 0 / 8  (A-frag K offset / 2, C-row half)
    const int b    = blockIdx.y;
    const int i0   = blockIdx.x * 128;

    // stage Q tile (transpose l-major global -> [i][l])
    for (int idx = tid; idx < 128 * 64; idx += 256) {
        int l = idx >> 7, i = idx & 127;
        sQ[i * 64 + l] = gmat[((size_t)b * LAYER + l) * N_SEQ + i0 + i];
    }

    f32x8 o_acc[16];
#pragma unroll
    for (int t = 0; t < 16; ++t)
        o_acc[t] = (f32x8){0.f, 0.f, 0.f, 0.f, 0.f, 0.f, 0.f, 0.f};
    float mrun[8], lrun[8];
#pragma unroll
    for (int r = 0; r < 8; ++r) { mrun[r] = -1e30f; lrun[r] = 0.f; }

    __syncthreads();

    for (int jt = 0; jt < N_SEQ / 128; ++jt) {
        const int j0 = jt * 128;

        // stage K tile [j][l]
        for (int idx = tid; idx < 128 * 64; idx += 256) {
            int l = idx >> 7, j = idx & 127;
            sKV[j * 64 + l] = fmat[((size_t)b * LAYER + l) * N_SEQ + j0 + j];
        }
        // prefetch next j-tile's K data (64 rows x 128 bf16 = 64 cachelines;
        // one lead thread per row) -> global_prefetch_b8, no counter cost
        if (jt + 1 < N_SEQ / 128 && tid < 64) {
            __builtin_prefetch(
                &fmat[((size_t)b * LAYER + tid) * N_SEQ + j0 + 128], 0, 3);
        }
        __syncthreads();

        // ---- S tile: 16 WMMAs (8 col tiles x K=64 in 2 steps) -------------
        f32x8 st[8];
#pragma unroll
        for (int t = 0; t < 8; ++t)
            st[t] = (f32x8){0.f, 0.f, 0.f, 0.f, 0.f, 0.f, 0.f, 0.f};
#pragma unroll
        for (int kb = 0; kb < 64; kb += 32) {
            bf16x16 aq = ld_afrag(&sQ[(w * 16 + l4) * 64], kb, hi8);
#pragma unroll
            for (int t = 0; t < 8; ++t) {
                bf16x16 bk = *(const bf16x16*)&sKV[(t * 16 + l4) * 64 + kb + hi16];
                st[t] = wmma_bf16(aq, bk, st[t]);
            }
        }

        // ---- online softmax (C layout: elem r = row r+hi8, col = l4) ------
        float mnew[8], corr[8];
#pragma unroll
        for (int r = 0; r < 8; ++r) {
            float v = st[0][r];
#pragma unroll
            for (int t = 1; t < 8; ++t) v = fmaxf(v, st[t][r]);
            v = redmax16(v);
            mnew[r] = fmaxf(mrun[r], v);
            corr[r] = __expf(mrun[r] - mnew[r]);
            mrun[r] = mnew[r];
        }
#pragma unroll
        for (int t = 0; t < 8; ++t)
#pragma unroll
            for (int r = 0; r < 8; ++r) st[t][r] = __expf(st[t][r] - mnew[r]);
#pragma unroll
        for (int r = 0; r < 8; ++r) {
            float s = 0.f;
#pragma unroll
            for (int t = 0; t < 8; ++t) s += st[t][r];
            s = redsum16(s);
            lrun[r] = lrun[r] * corr[r] + s;
        }
#pragma unroll
        for (int t = 0; t < 16; ++t)
#pragma unroll
            for (int r = 0; r < 8; ++r) o_acc[t][r] *= corr[r];

        // P -> LDS bf16 (wave-private rows)
#pragma unroll
        for (int t = 0; t < 8; ++t)
#pragma unroll
            for (int r = 0; r < 8; ++r)
                sP[(w * 16 + r + hi8) * 128 + t * 16 + l4] = (__bf16)st[t][r];
        __syncthreads();

        // P A-fragments, reused across all 4 value chunks
        bf16x16 af[4];
#pragma unroll
        for (int kk = 0; kk < 4; ++kk)
            af[kk] = ld_afrag(&sP[(w * 16 + l4) * 128], kk * 32, hi8);

        // ---- O += P * V : 64 WMMAs over 4 chunks of 64 channels -----------
        for (int cc = 0; cc < 4; ++cc) {
            for (int idx = tid; idx < 64 * 128; idx += 256) {
                int ci = idx >> 7, j = idx & 127;
                sKV[idx] = hmat[((size_t)b * DIM1 + cc * 64 + ci) * N_SEQ + j0 + j];
            }
            // prefetch next V chunk (or next tile's first chunk)
            if (tid < 64) {
                int ncc = cc + 1;
                size_t pj0 = (size_t)j0;
                if (ncc == 4) { ncc = 0; pj0 += 128; }
                if (pj0 < N_SEQ) {
                    __builtin_prefetch(
                        &hmat[((size_t)b * DIM1 + ncc * 64 + tid) * N_SEQ + pj0], 0, 3);
                }
            }
            __syncthreads();
#pragma unroll
            for (int t2 = 0; t2 < 4; ++t2) {
                const int ot = cc * 4 + t2;
#pragma unroll
                for (int kk = 0; kk < 4; ++kk) {
                    bf16x16 bv = *(const bf16x16*)&sKV[(t2 * 16 + l4) * 128 + kk * 32 + hi16];
                    o_acc[ot] = wmma_bf16(af[kk], bv, o_acc[ot]);
                }
            }
            __syncthreads();
        }
    }

    // ---- epilogue: O/l, out = gamma*O^T + q -------------------------------
    const float gmv = gamma[0];
    float rl[8];
#pragma unroll
    for (int r = 0; r < 8; ++r) rl[r] = 1.0f / lrun[r];
#pragma unroll
    for (int t = 0; t < 16; ++t) {
#pragma unroll
        for (int r = 0; r < 8; ++r) {
            int c   = (t >> 2) * 64 + (t & 3) * 16 + l4;
            int row = i0 + w * 16 + r + hi8;
            size_t idx = ((size_t)b * DIM1 + c) * N_SEQ + row;
            outp[idx] = gmv * (o_acc[t][r] * rl[r]) + qres[idx];
        }
    }
}

// ---------------------------------------------------------------------------
extern "C" void kernel_launch(void* const* d_in, const int* in_sizes, int n_in,
                              void* d_out, int out_size, void* d_ws, size_t ws_size,
                              hipStream_t stream)
{
    const float* q   = (const float*)d_in[0];
    const float* kin = (const float*)d_in[1];
    const float* Wf  = (const float*)d_in[2];
    const float* bf_ = (const float*)d_in[3];
    const float* fs  = (const float*)d_in[4];
    const float* fb  = (const float*)d_in[5];
    const float* fm  = (const float*)d_in[6];
    const float* fv  = (const float*)d_in[7];
    const float* Wg  = (const float*)d_in[8];
    const float* bg  = (const float*)d_in[9];
    const float* gs  = (const float*)d_in[10];
    const float* gb  = (const float*)d_in[11];
    const float* gmn = (const float*)d_in[12];
    const float* gv  = (const float*)d_in[13];
    const float* Wh  = (const float*)d_in[14];
    const float* bh  = (const float*)d_in[15];
    const float* hs  = (const float*)d_in[16];
    const float* hb  = (const float*)d_in[17];
    const float* hm  = (const float*)d_in[18];
    const float* hv  = (const float*)d_in[19];
    const float* gamma = (const float*)d_in[20];
    float* outp = (float*)d_out;

    __bf16* f_bf = (__bf16*)d_ws;                                  // 4 MB
    __bf16* g_bf = f_bf + (size_t)BATCH * LAYER * N_SEQ;           // 4 MB
    __bf16* h_bf = g_bf + (size_t)BATCH * LAYER * N_SEQ;           // 16 MB

    dim3 blk(256);
    convbngelu_kernel<<<dim3((LAYER / 4) * 16, BATCH), blk, 0, stream>>>(
        q, Wf, bf_, fs, fb, fm, fv, f_bf, DIM1, LAYER);
    convbngelu_kernel<<<dim3((LAYER / 4) * 16, BATCH), blk, 0, stream>>>(
        kin, Wg, bg, gs, gb, gmn, gv, g_bf, DIM2, LAYER);
    convbngelu_kernel<<<dim3((DIM1 / 4) * 16, BATCH), blk, 0, stream>>>(
        kin, Wh, bh, hs, hb, hm, hv, h_bf, DIM2, DIM1);

    flash_attn_kernel<<<dim3(N_SEQ / 128, BATCH), blk, 0, stream>>>(
        f_bf, g_bf, h_bf, q, gamma, outp);
}
